// ooi_net_27238682591291
// MI455X (gfx1250) — compile-verified, matrix operand order checked by reading, not run on hardware
//
#include <hip/hip_runtime.h>
#include <hip/hip_bf16.h>
#include <stdint.h>
#include <stddef.h>

// ---------------------------------------------------------------------------
// Problem constants (match the JAX reference)
// ---------------------------------------------------------------------------
#define BATCH   64
#define NNODE   128
#define NEDGE   4096
#define NPAIR   2048
#define DIN     512
#define DGC     256
#define EFEAT   256
#define CIN     512          // DGC + EFEAT
#define HIDN    512
#define OUTTOT  25

typedef __bf16 bf16_t;
typedef __attribute__((ext_vector_type(16))) __bf16 v16bf;
typedef __attribute__((ext_vector_type(8)))  __bf16 v8bf;
typedef __attribute__((ext_vector_type(8)))  float  v8f;
typedef unsigned int __attribute__((ext_vector_type(4))) u32x4;
typedef int          __attribute__((ext_vector_type(4))) i32x4;
typedef int          __attribute__((ext_vector_type(8))) i32x8;

union FragU { v16bf v; v8bf h[2]; };

// Tell the compiler that `p` (LDS staged by the TDM, which LLVM cannot see
// writing memory) now holds defined data: the empty asm may have written
// through p, so later ds_loads cannot be folded to undef / hoisted.
#define LDS_PUBLISH(p) asm volatile("" : : "v"((const void*)(p)) : "memory")

// Load a 16x32 bf16 WMMA operand fragment (A layout; B uses the same pattern
// with rows = output columns since weights are stored transposed [N][K]).
// Per CDNA5 ISA 7.12.2: lane L -> row = rowBase + (L&15), half = L>>4;
// elements 0..7  = K[k0 + half*8     .. +8)   (one 16-byte load)
// elements 8..15 = K[k0 + 16+half*8  .. +8)   (one 16-byte load)
__device__ __forceinline__ v16bf load_frag(const bf16_t* __restrict__ base,
                                           int ld, int rowBase, int k0, int lane) {
  const int r    = rowBase + (lane & 15);
  const int half = lane >> 4;
  const bf16_t* p = base + (size_t)r * ld + k0 + half * 8;
  FragU f;
  f.h[0] = *(const v8bf*)(p);
  f.h[1] = *(const v8bf*)(p + 16);
  return f.v;
}

__device__ __forceinline__ v8f wmma_bf16(v16bf a, v16bf b, v8f c) {
  return __builtin_amdgcn_wmma_f32_16x16x32_bf16(false, a, false, b,
                                                 (short)0, c, false, false);
}

// ---------------------------------------------------------------------------
// TDM: DMA a [rows x K] bf16 tile from global into LDS offset 0 with an
// 8-element (16B) pad appended per row, matching the conflict-free K+8 LDS
// stride. 2-D tile -> groups 2/3 unused (tile_dim2 = 0). One issue per block.
// ---------------------------------------------------------------------------
__device__ __forceinline__ void tdm_load_tile_bf16(const bf16_t* gptr,
                                                   int K, int rows) {
  const uint64_t ga = (uint64_t)(uintptr_t)gptr;
  // pad every row: row bytes = K*2 = (1<<iv)*8  ->  iv = log2(K) - 2
  const int iv = (K == 512) ? 7 : 6;      // 1024B / 512B row period
  const int pa = 3;                       // pad_amount: 3 -> 4 dwords (16B)

  u32x4 g0;
  g0.x = 1u;                                            // count=1, user mode
  g0.y = 0u;                                            // lds_addr = 0
  g0.z = (unsigned)(ga & 0xffffffffu);                  // global_addr[31:0]
  g0.w = (unsigned)((ga >> 32) & 0x01ffffffu) | (2u << 30); // addr[56:32], type=2

  i32x8 g1;
  g1.s0 = (1 << 16)            // data_size = 1 (2 bytes)
        | (1 << 20)            // pad_enable
        | (iv << 22)           // pad_interval
        | (pa << 25);          // pad_amount
  g1.s1 = (K & 0xffff) << 16;                 // tensor_dim0[15:0]  @ bits 63:48
  g1.s2 = ((K >> 16) & 0xffff)                // tensor_dim0[31:16]
        | ((rows & 0xffff) << 16);            // tensor_dim1[15:0]  @ bits 95:80
  g1.s3 = ((K & 0xffff) << 16);               // tensor_dim1[31:16]=0, tile_dim0=K
  g1.s4 = rows & 0xffff;                      // tile_dim1 = rows, tile_dim2 = 0
  g1.s5 = K;                                  // tensor_dim0_stride[31:0]
  g1.s6 = 0;
  g1.s7 = 0;

  i32x4 z4 = {0, 0, 0, 0};
#if __clang_major__ >= 23
  i32x8 z8 = {0, 0, 0, 0, 0, 0, 0, 0};
  __builtin_amdgcn_tensor_load_to_lds(g0, g1, z4, z4, z8, 0);
#else
  __builtin_amdgcn_tensor_load_to_lds(g0, g1, z4, z4, 0);
#endif
}

// ---------------------------------------------------------------------------
// Weight prep: f32 [K][N] -> bf16 [Npad][K]  (transposed, zero padded rows)
// ---------------------------------------------------------------------------
__global__ __launch_bounds__(256) void transpose_to_bf16(
    const float* __restrict__ src, bf16_t* __restrict__ dst,
    int K, int Ncols, int Npad) {
  int idx = blockIdx.x * 256 + threadIdx.x;
  if (idx >= Npad * K) return;
  int n = idx / K, k = idx - n * K;
  dst[(size_t)n * K + k] = (n < Ncols) ? (bf16_t)src[(size_t)k * Ncols + n]
                                       : (bf16_t)0.0f;
}

// ---------------------------------------------------------------------------
// GCN neighbor aggregation:  out[b,n,:] = x[b,n,:] + sum_{e: dst[e]==n} x[b,src[e],:]
// One block per (graph, node). Deterministic (no atomics): scan the edge list
// staged through LDS; dst compare is wave-uniform -> scalar branch.
// ---------------------------------------------------------------------------
__global__ __launch_bounds__(256) void gcn_aggregate(
    const float* __restrict__ x,    // [B, NNODE, D] f32
    const int*   __restrict__ ei,   // [B, 2, NEDGE]
    bf16_t*      __restrict__ out,  // [B*NNODE, D] bf16 (GEMM A layout)
    int D) {
  const int b   = blockIdx.x;
  const int n   = blockIdx.y;
  const int tid = threadIdx.x;
  const int nf  = D >> 8;                 // 1 or 2 features per thread

  __shared__ int s_src[256];
  __shared__ int s_dst[256];

  float acc[2];
  const float* xn = x + ((size_t)b * NNODE + n) * D;
  for (int i = 0; i < nf; ++i) acc[i] = xn[tid + (i << 8)];

  const int* src = ei + (size_t)b * 2 * NEDGE;
  const int* dst = src + NEDGE;

  for (int e0 = 0; e0 < NEDGE; e0 += 256) {
    s_src[tid] = src[e0 + tid];
    s_dst[tid] = dst[e0 + tid];
    __syncthreads();
    for (int j = 0; j < 256; ++j) {
      if (s_dst[j] == n) {                // uniform across block
        const float* xs = x + ((size_t)b * NNODE + s_src[j]) * D;
        for (int i = 0; i < nf; ++i) acc[i] += xs[tid + (i << 8)];
      }
    }
    __syncthreads();
  }
  bf16_t* o = out + ((size_t)b * NNODE + n) * D;
  for (int i = 0; i < nf; ++i) o[tid + (i << 8)] = (bf16_t)acc[i];
}

// ---------------------------------------------------------------------------
// C[M][256] = relu(A[M][K](bf16) @ W[K][256]),  W given transposed bf16 [256][K].
// Block: 256 thr = 8 waves; tile = 32 rows x 256 cols (32 cols / wave,
// 2 M-tiles -> each weight fragment feeds two WMMAs).
// A tile staged via TDM tensor_load_to_lds (row pad 16B -> conflict-free
// ds_load_b128 fragments); weights streamed from L2 as b128 pairs.
// ---------------------------------------------------------------------------
__global__ __launch_bounds__(256) void gemm_relu_bf16(
    const bf16_t* __restrict__ A,   // [M][K]
    const bf16_t* __restrict__ BT,  // [256][K]
    float*        __restrict__ C,   // [M][256]
    int K) {
  const int tid  = threadIdx.x;
  const int lane = tid & 31;
  const int wave = tid >> 5;
  const int rowBase = blockIdx.x * 32;
  const int ldA = K + 8;

  __shared__ __align__(16) bf16_t sA[32 * (DIN + 8)];   // only LDS symbol -> offset 0

  if (wave == 0) {                       // one TDM descriptor per block
    tdm_load_tile_bf16(A + (size_t)rowBase * K, K, 32);
    __builtin_amdgcn_s_wait_tensorcnt(0);
  }
  __syncthreads();
  LDS_PUBLISH(sA);                       // sA now holds DMA'd data: keep the loads

  const int nBase = wave * 32;
  v8f acc[2][2] = {};
  for (int k0 = 0; k0 < K; k0 += 32) {
    v16bf a0 = load_frag(sA, ldA, 0,  k0, lane);
    v16bf a1 = load_frag(sA, ldA, 16, k0, lane);
#pragma unroll
    for (int nt = 0; nt < 2; ++nt) {
      v16bf bfr = load_frag(BT, K, nBase + nt * 16, k0, lane);
      acc[0][nt] = wmma_bf16(a0, bfr, acc[0][nt]);
      acc[1][nt] = wmma_bf16(a1, bfr, acc[1][nt]);
    }
  }
  // ReLU + store f32. C layout: lane -> col (lane&15), vgpr r -> row r+8*(lane>>4).
  const int half = lane >> 4;
#pragma unroll
  for (int mt = 0; mt < 2; ++mt) {
#pragma unroll
    for (int nt = 0; nt < 2; ++nt) {
      int col = nBase + nt * 16 + (lane & 15);
#pragma unroll
      for (int r = 0; r < 8; ++r) {
        float v = acc[mt][nt][r];
        v = v > 0.0f ? v : 0.0f;
        C[(size_t)(rowBase + mt * 16 + half * 8 + r) * 256 + col] = v;
      }
    }
  }
}

// ---------------------------------------------------------------------------
// Fused pair-gather + 3-head MLP.
// One block per 32 rows of ci (block never straddles a graph: P % 32 == 0).
//   ci[r]  = [ node_emb[i0]+node_emb[i1] (256) | interaction[i0,i1] (256) ]
//   per head: hid = relu(ci @ W1 + b1); out = hid @ W2 + b2
// A single 32x(512+8) bf16 LDS tile is reused for ci -> hid each head (the
// rebuilt gather set is L2-resident, trading ~2x134MB of L2 re-reads for a
// ~6.3GB reduction in weight streaming). Layer1: 8 waves x 64 hid cols with
// 2 M-tiles sharing every weight fragment. Layer2: waves 0/1, W2 padded to 16.
// ---------------------------------------------------------------------------
struct MlpP {
  const float*  node_emb;   // [B*NNODE, 256] f32
  const int*    pairs;      // [B, NPAIR, 2]
  const float*  inter;      // [B, NNODE, NNODE, EFEAT] f32
  const bf16_t* W1T;        // [3][HIDN][CIN]  (transposed)
  const bf16_t* W2T;        // [3][16][HIDN]   (transposed, zero padded)
  const float*  b1[3];
  const float*  b2[3];
  float*        out;        // [B, NPAIR, 25] f32
};

__global__ __launch_bounds__(256) void mlp_fused(MlpP P) {
  const int tid  = threadIdx.x;
  const int lane = tid & 31;
  const int wave = tid >> 5;
  const int half = lane >> 4;
  const int rowBase = blockIdx.x * 32;          // global ci row
  const int bg = rowBase >> 11;                 // graph id (P=2048 rows/graph)

  const int OUTN[3] = {6, 5, 14};
  const int OFF[3]  = {0, 6, 11};

  __shared__ __align__(16) bf16_t sT[32 * (CIN + 8)];   // ci, then hid (reused)
  __shared__ int sI0[32], sI1[32];
  const int ldT = CIN + 8;

  if (tid < 32) {
    int g = rowBase + tid;
    int p = g & (NPAIR - 1);
    const int* pr = P.pairs + ((size_t)bg * NPAIR + p) * 2;
    sI0[tid] = pr[0];
    sI1[tid] = pr[1];
  }
  __syncthreads();

  const int nBase = wave * 64;                  // hid columns owned by this wave

  for (int h = 0; h < 3; ++h) {
    // ---- (Re)build ci tile in LDS (bf16); gathers hit L2 after head 0 ----
    for (int idx = tid; idx < 32 * CIN; idx += 256) {
      int r = idx >> 9;                  // / 512
      int c = idx & (CIN - 1);
      int i0 = sI0[r], i1 = sI1[r];
      float v;
      if (c < DGC) {
        const float* e0 = P.node_emb + ((size_t)bg * NNODE + i0) * DGC;
        const float* e1 = P.node_emb + ((size_t)bg * NNODE + i1) * DGC;
        v = e0[c] + e1[c];
      } else {
        v = P.inter[(((size_t)bg * NNODE + i0) * NNODE + i1) * EFEAT + (c - DGC)];
      }
      sT[r * ldT + c] = (bf16_t)v;
    }
    __syncthreads();

    const bf16_t* W1 = P.W1T + (size_t)h * HIDN * CIN;
    const float*  b1 = P.b1[h];

    // ---- Layer 1: hid = relu(ci @ W1 + b1), 32 x 512 ----
    v8f acc[2][4] = {};
    for (int k0 = 0; k0 < CIN; k0 += 32) {
      v16bf a0 = load_frag(sT, ldT, 0,  k0, lane);
      v16bf a1 = load_frag(sT, ldT, 16, k0, lane);
      if (k0 + 32 < CIN)   // keep next weight k-panel flowing through L2
        __builtin_prefetch(W1 + (size_t)(nBase + (lane & 15)) * CIN + k0 + 32, 0, 0);
#pragma unroll
      for (int nt = 0; nt < 4; ++nt) {
        v16bf bfr = load_frag(W1, CIN, nBase + nt * 16, k0, lane);
        acc[0][nt] = wmma_bf16(a0, bfr, acc[0][nt]);
        acc[1][nt] = wmma_bf16(a1, bfr, acc[1][nt]);
      }
    }
    __syncthreads();                     // everyone done reading ci

    // ---- bias + ReLU -> hid tile (overwrites ci) ----
#pragma unroll
    for (int mt = 0; mt < 2; ++mt) {
#pragma unroll
      for (int nt = 0; nt < 4; ++nt) {
        int col = nBase + nt * 16 + (lane & 15);
        float bias = b1[col];
#pragma unroll
        for (int r = 0; r < 8; ++r) {
          float v = acc[mt][nt][r] + bias;
          v = v > 0.0f ? v : 0.0f;
          sT[(mt * 16 + half * 8 + r) * ldT + col] = (bf16_t)v;
        }
      }
    }
    __syncthreads();

    // ---- Layer 2: out = hid @ W2 + b2  (<=14 cols, one 16-wide tile) ----
    if (wave < 2) {                      // wave w handles M-tile w
      const bf16_t* W2 = P.W2T + (size_t)h * 16 * HIDN;
      v8f o = {};
      for (int k0 = 0; k0 < HIDN; k0 += 32) {
        v16bf a = load_frag(sT, ldT, wave * 16, k0, lane);
        v16bf bfr = load_frag(W2, HIDN, 0, k0, lane);
        o = wmma_bf16(a, bfr, o);
      }
      int n = lane & 15;
      if (n < OUTN[h]) {
        float bias = P.b2[h][n];
#pragma unroll
        for (int r = 0; r < 8; ++r) {
          int g = rowBase + wave * 16 + half * 8 + r;
          P.out[(size_t)g * OUTTOT + OFF[h] + n] = o[r] + bias;
        }
      }
    }
    __syncthreads();   // all waves: sT rebuilt as ci next head
  }
}

// ---------------------------------------------------------------------------
// Host-side orchestration
// ---------------------------------------------------------------------------
extern "C" void kernel_launch(void* const* d_in, const int* in_sizes, int n_in,
                              void* d_out, int out_size, void* d_ws, size_t ws_size,
                              hipStream_t stream) {
  (void)in_sizes; (void)n_in; (void)out_size; (void)ws_size;

  const float* obj   = (const float*)d_in[0];   // [64,128,512]
  const int*   ei    = (const int*)  d_in[1];   // [64,2,4096]
  const int*   pairs = (const int*)  d_in[2];   // [64,2048,2]
  const float* inter = (const float*)d_in[3];   // [64,128,128,256]
  const float* Wg1   = (const float*)d_in[4];   // [512,256]
  const float* Wg2   = (const float*)d_in[5];   // [256,256]

  // Workspace carve-out (~31 MB total).
  char* w = (char*)d_ws;
  auto carve = [&](size_t bytes) {
    void* p = (void*)w;
    w += (bytes + 255) & ~(size_t)255;
    return p;
  };
  const size_t MROWS = (size_t)BATCH * NNODE;   // 8192
  bf16_t* m1   = (bf16_t*)carve(MROWS * DIN * 2);       // agg1 out (bf16)
  float*  hbuf = (float*) carve(MROWS * DGC * 4);       // gcn layer1 out (f32)
  bf16_t* m2   = (bf16_t*)carve(MROWS * DGC * 2);       // agg2 out (bf16)
  float*  ne   = (float*) carve(MROWS * DGC * 4);       // node_emb (f32)
  bf16_t* Wg1T = (bf16_t*)carve((size_t)DGC * DIN * 2);
  bf16_t* Wg2T = (bf16_t*)carve((size_t)DGC * DGC * 2);
  bf16_t* W1T  = (bf16_t*)carve((size_t)3 * HIDN * CIN * 2);
  bf16_t* W2T  = (bf16_t*)carve((size_t)3 * 16 * HIDN * 2);

  auto xpose = [&](const float* s, bf16_t* d, int K, int N, int Npad) {
    int total = Npad * K;
    transpose_to_bf16<<<(total + 255) / 256, 256, 0, stream>>>(s, d, K, N, Npad);
  };

  // Weight prep.
  xpose(Wg1, Wg1T, DIN, DGC, DGC);
  xpose(Wg2, Wg2T, DGC, DGC, DGC);
  const int OUTN[3] = {6, 5, 14};
  for (int h = 0; h < 3; ++h) {
    xpose((const float*)d_in[6 + 4 * h], W1T + (size_t)h * HIDN * CIN, CIN, HIDN, HIDN);
    xpose((const float*)d_in[8 + 4 * h], W2T + (size_t)h * 16 * HIDN, HIDN, OUTN[h], 16);
  }

  // GCN layer 1: aggregate then relu((m+x) @ W1) via WMMA.
  gcn_aggregate<<<dim3(BATCH, NNODE), 256, 0, stream>>>(obj, ei, m1, DIN);
  gemm_relu_bf16<<<MROWS / 32, 256, 0, stream>>>(m1, Wg1T, hbuf, DIN);

  // GCN layer 2.
  gcn_aggregate<<<dim3(BATCH, NNODE), 256, 0, stream>>>(hbuf, ei, m2, DGC);
  gemm_relu_bf16<<<MROWS / 32, 256, 0, stream>>>(m2, Wg2T, ne, DGC);

  // Fused pair-gather + 3-head MLP.
  MlpP P;
  P.node_emb = ne;
  P.pairs    = pairs;
  P.inter    = inter;
  P.W1T      = W1T;
  P.W2T      = W2T;
  P.b1[0] = (const float*)d_in[7];  P.b1[1] = (const float*)d_in[11]; P.b1[2] = (const float*)d_in[15];
  P.b2[0] = (const float*)d_in[9];  P.b2[1] = (const float*)d_in[13]; P.b2[2] = (const float*)d_in[17];
  P.out = (float*)d_out;
  const size_t CIROWS = (size_t)BATCH * NPAIR;  // 131072
  mlp_fused<<<CIROWS / 32, 256, 0, stream>>>(P);
}
